// GatedShift_14156212208116
// MI455X (gfx1250) — compile-verified, matrix operand order checked by reading, not run on hardware
//
#include <hip/hip_runtime.h>
#include <math.h>

typedef __attribute__((ext_vector_type(16))) _Float16 v16h;
typedef __attribute__((ext_vector_type(8)))  _Float16 v8h;
typedef __attribute__((ext_vector_type(8)))  float    v8f;
typedef __attribute__((ext_vector_type(4)))  float    vf4;

#define HW      3136                    // 56*56
#define NTSTR   (256 * 3136)            // stride of one nt frame in x/out (floats)
#define ACT_OFF 802816                  // gate floats before act region in d_ws

// Padded act tensor: [clip(8), frame(18), row(58), col(58), ch(64)] f16
#define PFRM    (3364 * 64)             // 58*58*64  f16 per padded frame
#define PCLIP   (18 * PFRM)             // f16 per padded clip
#define ACT2_F16 ((size_t)8 * PCLIP)    // 31,002,624 f16
#define SLACK_F16 32768                 // absorbs discarded-row overreads
#define ACT2_TOTAL (ACT2_F16 + SLACK_F16)   // multiple of 8

// ---------------------------------------------------------------------------
// Kernel Z: zero the padded act tensor (halos become conv zero-padding).
// ---------------------------------------------------------------------------
__global__ __launch_bounds__(256) void gsm_zeroact(_Float16* __restrict__ act)
{
    const size_t i = ((size_t)blockIdx.x * blockDim.x + threadIdx.x) * 8;
    if (i >= ACT2_TOTAL) return;
    v8h z = {};
    __builtin_nontemporal_store(z, (v8h*)(act + i));
}

// ---------------------------------------------------------------------------
// Kernel 0: act[clip, t+1, h+1, w+1, c] = relu(bn(x[:, :64]))  (f16, padded)
//   BN applied once here instead of 27x (tap-amplified) in the conv.
//   LDS tile transpose: coalesced f32 reads, coalesced 16B f16 writes.
// ---------------------------------------------------------------------------
__global__ __launch_bounds__(256) void gsm_bnact(
    const float* __restrict__ x,
    const float* __restrict__ gamma, const float* __restrict__ beta,
    const float* __restrict__ mean,  const float* __restrict__ var,
    _Float16* __restrict__ act)
{
    __shared__ float tile[64][65];      // 65: conflict-free transpose
    __shared__ float s_scale[64];
    __shared__ float s_shift[64];

    if (threadIdx.x < 64) {
        const int c = threadIdx.x;
        const float sc = gamma[c] * rsqrtf(var[c] + 1e-5f);
        s_scale[c] = sc;
        s_shift[c] = beta[c] - mean[c] * sc;
    }
    __syncthreads();

    const int tw  = blockIdx.x % 49;    // 49 tiles of 64 cover HW=3136
    const int nt  = blockIdx.x / 49;
    const int hw0 = tw * 64;
    const int s   = threadIdx.x & 63;
    const int cg  = threadIdx.x >> 6;   // 0..3

    #pragma unroll
    for (int i = 0; i < 16; ++i) {
        const int c = cg * 16 + i;
        const float xv = x[(size_t)nt * NTSTR + (size_t)c * HW + hw0 + s];
        tile[s][c] = fmaxf(fmaf(xv, s_scale[c], s_shift[c]), 0.0f);
    }
    __syncthreads();

    const int clip = nt >> 4;
    const int t    = nt & 15;
    const int c0    = (threadIdx.x & 7) * 8;
    const int sbase = threadIdx.x >> 3;  // 0..31
    #pragma unroll
    for (int it = 0; it < 2; ++it) {
        const int ss = sbase + it * 32;
        const int hw = hw0 + ss;
        const int hh = hw / 56;
        const int wwp = hw - hh * 56;
        v8h hv;
        #pragma unroll
        for (int j = 0; j < 8; ++j) hv[j] = (_Float16)tile[ss][c0 + j];
        _Float16* dst = act + (size_t)clip * PCLIP + (size_t)(t + 1) * PFRM
                            + ((size_t)(hh + 1) * 58 + (wwp + 1)) * 64 + c0;
        *(v8h*)dst = hv;                 // b128 store
    }
}

// ---------------------------------------------------------------------------
// Kernel 1: gate = tanh(conv3d(act)) via v_wmma_f32_16x16x32_f16
//   grid  : 8(n) * 16(t) * 7(h-blocks of 8) = 896 blocks, 8 waves/block
//   D tile: M = 16 w-positions, N cols 0/1 = conv groups, K = 32-ch halves.
//   Halo-padded act -> A frags are global_load_b128 at immediate offsets
//   from one per-wb base (no boundary VALU). B frags are ds_load_b128 at
//   immediate offsets; zero columns via a one-time pointer select onto a
//   zero mirror. __syncthreads() per wb pins B frags in LDS (no 400-VGPR
//   LICM blow-up); single live accumulator.
// ---------------------------------------------------------------------------
__global__ __launch_bounds__(256, 3) void gsm_gate_wmma(
    const _Float16* __restrict__ act,
    const float* __restrict__ cw, const float* __restrict__ cb,
    float* __restrict__ gws)
{
    // [tap][kk][khi][16 f16] in exact per-lane B-register order, + zero mirror
    __shared__ __align__(32) _Float16 s_bpack[27 * 64];
    __shared__ __align__(32) _Float16 s_bzpack[27 * 64];

    for (int idx = threadIdx.x; idx < 1728; idx += 256) {
        const int e    = idx & 15;
        const int bkhi = (idx >> 4) & 1;
        const int bkk  = (idx >> 5) & 1;
        const int tap  = idx >> 6;
        const int klocal = (e < 8) ? (bkhi * 8 + e) : (16 + bkhi * 8 + (e - 8));
        s_bpack[idx]  = (_Float16)cw[(bkk * 32 + klocal) * 27 + tap];
        s_bzpack[idx] = (_Float16)0.0f;
    }
    __syncthreads();

    const int wave = threadIdx.x >> 5;
    const int lane = threadIdx.x & 31;
    const int m    = lane & 15;          // M row (w position within 16-block)
    const int khi  = lane >> 4;
    const int khi8 = khi << 3;

    int b = blockIdx.x;
    const int hb = b % 7;  b /= 7;
    const int t  = b & 15; b >>= 4;
    const int n  = b;                    // clip 0..7
    const int h  = hb * 8 + wave;        // 0..55

    // One-time zero-column selection for B fragments (2 cndmasks total)
    const _Float16* b0 = ((lane & 15) == 0 ? s_bpack : s_bzpack) + khi * 16;
    const _Float16* b1 = ((lane & 15) == 1 ? s_bpack : s_bzpack) + khi * 16 + 32;

    // Per-lane A base: padded frame t+kt, padded row h+kh, padded col wb*16+m+kw
    const _Float16* abase = act + (size_t)n * PCLIP + (size_t)t * PFRM
                                + ((size_t)h * 58 + m) * 64 + khi8;

    #pragma unroll 1
    for (int wb = 0; wb < 4; ++wb) {     // 4 w-blocks of 16 cover W=56
        __syncthreads();                 // pin B frags in LDS across wb iters
        const _Float16* ap = abase + wb * (16 * 64);
        v8f acc = {};

        #pragma unroll
        for (int kt = 0; kt < 3; ++kt) {
            #pragma unroll
            for (int kh = 0; kh < 3; ++kh) {
                #pragma unroll
                for (int kw = 0; kw < 3; ++kw) {
                    const int tap  = (kt * 3 + kh) * 3 + kw;
                    const int aoff = kt * PFRM + kh * (58 * 64) + kw * 64;
                    #pragma unroll
                    for (int kk = 0; kk < 2; ++kk) {   // K half == conv group
                        const _Float16* bp = (kk == 0 ? b0 : b1) + tap * 64;
                        const v16h bf = *(const v16h*)bp;       // 2x ds b128

                        const _Float16* a = ap + aoff + kk * 32;
                        const v8h lo = *(const v8h*)a;          // global b128
                        const v8h hi = *(const v8h*)(a + 16);   // global b128
                        const v16h af = __builtin_shufflevector(lo, hi,
                            0, 1, 2, 3, 4, 5, 6, 7,
                            8, 9, 10, 11, 12, 13, 14, 15);

                        acc = __builtin_amdgcn_wmma_f32_16x16x32_f16(
                            false, af, false, bf, (short)0, acc, false, false);
                    }
                }
            }
        }

        // Epilogue: D column n lives in lanes n and n+16; VGPR r = row r (+8 hi)
        const int g = lane & 15;
        if (g < 2) {
            const float bias = cb[g];
            float* grow = gws + ((size_t)(n * 2 + g) * 16 + t) * HW
                              + (size_t)h * 56;
            #pragma unroll
            for (int r = 0; r < 8; ++r) {
                const int wpos = wb * 16 + r + khi8;
                if (wpos < 56) grow[wpos] = tanhf(acc[r] + bias);
            }
        }
    }
}

// ---------------------------------------------------------------------------
// Kernel 2: gated temporal shift + channel regroup for output channels 0..63
//   out[oc,t] = g[t+/-1]*x_src[t+/-1] + (1-g[t])*x_src[t], float4 vectorized
// ---------------------------------------------------------------------------
__global__ __launch_bounds__(256) void gsm_shift(
    const float* __restrict__ x, const float* __restrict__ gws,
    float* __restrict__ out)
{
    const long long tid = (long long)blockIdx.x * blockDim.x + threadIdx.x;
    const long long total = 128LL * 64 * 784;          // float4 units
    if (tid >= total) return;
    const int hw4 = (int)(tid % 784);
    int tmp = (int)(tid / 784);
    const int oc = tmp & 63;
    const int nt = tmp >> 6;
    const int n = nt >> 4, t = nt & 15;
    const int g   = oc >> 5;
    const int ocl = oc & 31;
    const int src = ((ocl & 1) << 4) + (ocl >> 1);     // undo channel regroup
    const int xch = (g << 5) + src;
    const size_t pos = (size_t)hw4 * 4;

    const vf4 xc = *(const vf4*)(x   + (size_t)nt * NTSTR + (size_t)xch * HW + pos);
    const vf4 gc = *(const vf4*)(gws + ((size_t)(n * 2 + g) * 16 + t) * HW + pos);
    const int t2 = g ? (t - 1) : (t + 1);              // right / left shift
    vf4 o;
    if ((unsigned)t2 < 16u) {
        const int nt2 = (n << 4) + t2;
        const vf4 x2 = *(const vf4*)(x   + (size_t)nt2 * NTSTR + (size_t)xch * HW + pos);
        const vf4 g2 = *(const vf4*)(gws + ((size_t)(n * 2 + g) * 16 + t2) * HW + pos);
        #pragma unroll
        for (int i = 0; i < 4; ++i)
            o[i] = g2[i] * x2[i] + (1.0f - gc[i]) * xc[i];
    } else {
        #pragma unroll
        for (int i = 0; i < 4; ++i)
            o[i] = (1.0f - gc[i]) * xc[i];
    }
    *(vf4*)(out + (size_t)nt * NTSTR + (size_t)oc * HW + pos) = o;
}

// ---------------------------------------------------------------------------
// Kernel 3: passthrough copy of channels 64..255 (75% of traffic),
// non-temporal so it does not evict the L2-resident act tensor.
// ---------------------------------------------------------------------------
__global__ __launch_bounds__(256) void gsm_copy(
    const float* __restrict__ x, float* __restrict__ out)
{
    const long long tid = (long long)blockIdx.x * blockDim.x + threadIdx.x;
    const long long total = 128LL * 192 * 784;         // float4 units
    if (tid >= total) return;
    const int per = 192 * 784;                         // 150528 float4 per frame
    const int chunk = (int)(tid / per);
    const int rem   = (int)(tid % per);
    const size_t off = (size_t)chunk * NTSTR + (size_t)64 * HW + (size_t)rem * 4;
    vf4 v = __builtin_nontemporal_load((const vf4*)(x + off));
    __builtin_nontemporal_store(v, (vf4*)(out + off));
}

// ---------------------------------------------------------------------------
extern "C" void kernel_launch(void* const* d_in, const int* in_sizes, int n_in,
                              void* d_out, int out_size, void* d_ws, size_t ws_size,
                              hipStream_t stream) {
    const float* x     = (const float*)d_in[0];
    const float* gamma = (const float*)d_in[1];
    const float* beta  = (const float*)d_in[2];
    const float* mean  = (const float*)d_in[3];
    const float* var   = (const float*)d_in[4];
    const float* cw    = (const float*)d_in[5];
    const float* cb    = (const float*)d_in[6];
    float* out = (float*)d_out;

    float*     gws = (float*)d_ws;                         // 3.2 MB gate
    _Float16*  act = (_Float16*)((float*)d_ws + ACT_OFF);  // ~62 MB padded act

    const unsigned zblocks = (unsigned)((ACT2_TOTAL / 8 + 255) / 256);
    gsm_zeroact<<<zblocks, 256, 0, stream>>>(act);

    gsm_bnact<<<dim3(128 * 49), 256, 0, stream>>>(x, gamma, beta, mean, var, act);

    gsm_gate_wmma<<<dim3(8 * 16 * 7), 256, 0, stream>>>(act, cw, cb, gws);

    const long long shift_total = 128LL * 64 * 784;
    gsm_shift<<<(unsigned)((shift_total + 255) / 256), 256, 0, stream>>>(x, gws, out);

    const long long copy_total = 128LL * 192 * 784;
    gsm_copy<<<(unsigned)((copy_total + 255) / 256), 256, 0, stream>>>(x, out);

    (void)in_sizes; (void)n_in; (void)out_size; (void)ws_size;
}
